// RNN_80668075753593
// MI455X (gfx1250) — matrix operand order repaired and checked by reading, not compiled, not run on hardware
//
#include <hip/hip_runtime.h>
#include <hip/hip_bf16.h>
#include <math.h>

#define VOCAB  8192
#define HIDDEN 1024
#define BATCH  64
#define SEQ    256

typedef __attribute__((ext_vector_type(16))) __bf16 v16bf;
typedef __attribute__((ext_vector_type(8)))  __bf16 v8bf;
typedef __attribute__((ext_vector_type(8)))  float  v8f;

// ---------------------------------------------------------------------------
// round-to-nearest-even f32 -> bf16
static __device__ __forceinline__ unsigned short f32_to_bf16_rne(float f) {
    unsigned int u = __builtin_bit_cast(unsigned int, f);
    u += 0x7FFFu + ((u >> 16) & 1u);
    return (unsigned short)(u >> 16);
}

// A fragment (16x32 bf16, M x K): lane l holds row m0+(l&15);
// halves 0..7 = K[k0 + 8*(l>>4) .. +7], halves 8..15 = same + 16.
static __device__ __forceinline__ v16bf load_a_frag(const __bf16* base) {
    v8bf lo = *reinterpret_cast<const v8bf*>(base);
    v8bf hi = *reinterpret_cast<const v8bf*>(base + 16);
    return __builtin_shufflevector(lo, hi, 0,1,2,3,4,5,6,7,8,9,10,11,12,13,14,15);
}

// B fragment (32x16 bf16, K x N) from an N-major (transposed) weight matrix:
// lane l holds col n0+(l&15); halves 0..7 = K[k0 + 8*(l>>4) .. +7], halves 8..15 = same + 16.
static __device__ __forceinline__ v16bf load_b_frag(const __bf16* base) {
    v8bf lo = *reinterpret_cast<const v8bf*>(base);
    v8bf hi = *reinterpret_cast<const v8bf*>(base + 16);
    return __builtin_shufflevector(lo, hi, 0,1,2,3,4,5,6,7,8,9,10,11,12,13,14,15);
}

// ---------------------------------------------------------------------------
// Register-blocked (MT x NT tiles) software-pipelined bf16 GEMM core, f32 acc.
// sched_barrier(0) fences keep the k+1 fragment loads ABOVE the k-step WMMA
// pack so the scheduler cannot collapse the double buffer into load-use.
template<int MT, int NT>
static __device__ __forceinline__ void gemm_bf16_rb(
    const __bf16* __restrict__ aBase,   // &A[m0+lr][lhi*8]
    const __bf16* __restrict__ bBase,   // &BT[n0+lr][lhi*8]
    v8f (&acc)[MT][NT]) {
    const long long TS = 16LL * HIDDEN;   // one 16-row/col tile stride

    v16bf a[MT], b[NT];
    #pragma unroll
    for (int i = 0; i < MT; ++i) a[i] = load_a_frag(aBase + i * TS);
    #pragma unroll
    for (int j = 0; j < NT; ++j) b[j] = load_b_frag(bBase + j * TS);

    #pragma unroll 2
    for (int k0 = 0; k0 < HIDDEN - 32; k0 += 32) {
        v16bf a_n[MT], b_n[NT];
        #pragma unroll
        for (int i = 0; i < MT; ++i) a_n[i] = load_a_frag(aBase + i * TS + k0 + 32);
        #pragma unroll
        for (int j = 0; j < NT; ++j) b_n[j] = load_b_frag(bBase + j * TS + k0 + 32);
        __builtin_amdgcn_sched_barrier(0);   // loads stay above, WMMAs below
        #pragma unroll
        for (int i = 0; i < MT; ++i)
            #pragma unroll
            for (int j = 0; j < NT; ++j)
                acc[i][j] = __builtin_amdgcn_wmma_f32_16x16x32_bf16(
                    false, a[i], false, b[j], (short)0, acc[i][j], false, false);
        __builtin_amdgcn_sched_barrier(0);
        #pragma unroll
        for (int i = 0; i < MT; ++i) a[i] = a_n[i];
        #pragma unroll
        for (int j = 0; j < NT; ++j) b[j] = b_n[j];
    }
    #pragma unroll
    for (int i = 0; i < MT; ++i)
        #pragma unroll
        for (int j = 0; j < NT; ++j)
            acc[i][j] = __builtin_amdgcn_wmma_f32_16x16x32_bf16(
                false, a[i], false, b[j], (short)0, acc[i][j], false, false);
}

// ---------------------------------------------------------------------------
// Prep: fp32 (K x N, row-major) -> bf16 transposed (N x K, row-major)
__global__ void k_transpose_bf16(const float* __restrict__ src,
                                 unsigned short* __restrict__ dst,
                                 int K, int N) {
    long long idx = (long long)blockIdx.x * blockDim.x + threadIdx.x;
    long long total = (long long)K * N;
    if (idx >= total) return;
    int n = (int)(idx / K);
    int k = (int)(idx - (long long)n * K);
    dst[idx] = f32_to_bf16_rne(src[(long long)k * N + n]);
}

// Prep: initial hidden state fp32 -> bf16
__global__ void k_init_h(const float* __restrict__ status,
                         unsigned short* __restrict__ hbf) {
    int idx = blockIdx.x * blockDim.x + threadIdx.x;
    if (idx < BATCH * HIDDEN) hbf[idx] = f32_to_bf16_rne(status[idx]);
}

// ---------------------------------------------------------------------------
// Fused per-step kernel. Both halves depend only on h_t:
//   blocks 0..3   (32 waves):  h_{t+1} = tanh(h_t @ w_hh + w_xh[X[:,t]] + b_h)
//   blocks 4..35  (256 waves): Y[t-1]  = h_t @ w_ho + b_o   (skipped at t==0)
// Each wave owns a 32x64 output tile (2 M-tiles x 4 N-tiles).
__global__ void __launch_bounds__(256)
k_rnn_step(const unsigned short* __restrict__ h_in_u,   // h_t, 64x1024 bf16
           unsigned short*       __restrict__ h_out_u,  // h_{t+1}, 64x1024 bf16
           float*                __restrict__ h_out_f,  // h_final slot, 64x1024 f32
           const unsigned short* __restrict__ whhT_u,   // 1024(n) x 1024(k) bf16
           const unsigned short* __restrict__ whoT_u,   // 8192(n) x 1024(k) bf16
           const int*            __restrict__ X,        // 64x256 tokens
           const float*          __restrict__ w_xh,     // 8192x1024 f32
           const float*          __restrict__ b_h,      // 1024
           const float*          __restrict__ b_o,      // 8192
           float*                __restrict__ Y_prev,   // Y[t-1] base, or nullptr
           int t) {
    const __bf16* h_in = reinterpret_cast<const __bf16*>(h_in_u);

    const unsigned wave = threadIdx.x >> 5;
    const unsigned lane = threadIdx.x & 31u;
    const unsigned lhi  = lane >> 4;
    const unsigned lr   = lane & 15u;

    if (blockIdx.x < 4) {
        // ---- recurrence: M=64, N=1024, K=1024; 32 wave-jobs ----
        const __bf16* whhT = reinterpret_cast<const __bf16*>(whhT_u);
        const unsigned job = blockIdx.x * 8u + wave;          // 0..31
        const unsigned m0  = (job & 1u) * 32u;
        const unsigned n0  = (job >> 1) * 64u;

        v8f acc[2][4] = {};
        gemm_bf16_rb<2, 4>(h_in + (m0 + lr) * HIDDEN + lhi * 8u,
                           whhT + (n0 + lr) * HIDDEN + lhi * 8u, acc);

        #pragma unroll
        for (int mi = 0; mi < 2; ++mi)
            #pragma unroll
            for (int j = 0; j < 4; ++j) {
                const unsigned n = n0 + (unsigned)j * 16u + lr;
                const float bias = b_h[n];
                #pragma unroll
                for (int r = 0; r < 8; ++r) {
                    const unsigned m = m0 + (unsigned)mi * 16u + lhi * 8u + (unsigned)r;
                    const unsigned tok = (unsigned)X[m * SEQ + t];
                    float v = tanhf(acc[mi][j][r] + w_xh[(long long)tok * HIDDEN + n] + bias);
                    h_out_u[m * HIDDEN + n] = f32_to_bf16_rne(v);
                    h_out_f[m * HIDDEN + n] = v;
                }
            }
    } else {
        // ---- projection of previous step: M=64, N=8192, K=1024; 256 wave-jobs ----
        if (Y_prev == nullptr) return;
        const __bf16* whoT = reinterpret_cast<const __bf16*>(whoT_u);
        const unsigned job = (blockIdx.x - 4u) * 8u + wave;   // 0..255
        const unsigned m0  = (job & 1u) * 32u;
        const unsigned n0  = (job >> 1) * 64u;

        v8f acc[2][4] = {};
        gemm_bf16_rb<2, 4>(h_in + (m0 + lr) * HIDDEN + lhi * 8u,
                           whoT + (long long)(n0 + lr) * HIDDEN + lhi * 8u, acc);

        #pragma unroll
        for (int mi = 0; mi < 2; ++mi)
            #pragma unroll
            for (int j = 0; j < 4; ++j) {
                const unsigned n = n0 + (unsigned)j * 16u + lr;
                const float bias = b_o[n];
                #pragma unroll
                for (int r = 0; r < 8; ++r) {
                    const unsigned m = m0 + (unsigned)mi * 16u + lhi * 8u + (unsigned)r;
                    // streaming 512 MiB output: NT store keeps bf16 weights in L2
                    __builtin_nontemporal_store(acc[mi][j][r] + bias,
                                                &Y_prev[(long long)m * VOCAB + n]);
                }
            }
    }
}

// Tail projection: Y[SEQ-1] = h_SEQ @ w_ho + b_o.  256 wave-jobs.
__global__ void __launch_bounds__(256)
k_rnn_project(const unsigned short* __restrict__ h_u,
              const unsigned short* __restrict__ whoT_u,
              const float*          __restrict__ b_o,
              float*                __restrict__ Yt) {
    const __bf16* h    = reinterpret_cast<const __bf16*>(h_u);
    const __bf16* whoT = reinterpret_cast<const __bf16*>(whoT_u);

    const unsigned wave = threadIdx.x >> 5;
    const unsigned lane = threadIdx.x & 31u;
    const unsigned lhi  = lane >> 4;
    const unsigned lr   = lane & 15u;

    const unsigned job = blockIdx.x * 8u + wave;   // 0..255
    const unsigned m0  = (job & 1u) * 32u;
    const unsigned n0  = (job >> 1) * 64u;

    v8f acc[2][4] = {};
    gemm_bf16_rb<2, 4>(h    + (m0 + lr) * HIDDEN + lhi * 8u,
                       whoT + (long long)(n0 + lr) * HIDDEN + lhi * 8u, acc);

    #pragma unroll
    for (int mi = 0; mi < 2; ++mi)
        #pragma unroll
        for (int j = 0; j < 4; ++j) {
            const unsigned n = n0 + (unsigned)j * 16u + lr;
            const float bias = b_o[n];
            #pragma unroll
            for (int r = 0; r < 8; ++r) {
                const unsigned m = m0 + (unsigned)mi * 16u + lhi * 8u + (unsigned)r;
                __builtin_nontemporal_store(acc[mi][j][r] + bias,
                                            &Yt[(long long)m * VOCAB + n]);
            }
        }
}

// ---------------------------------------------------------------------------
extern "C" void kernel_launch(void* const* d_in, const int* in_sizes, int n_in,
                              void* d_out, int out_size, void* d_ws, size_t ws_size,
                              hipStream_t stream) {
    const int*   X      = (const int*)  d_in[0];  // (64,256)
    const float* status = (const float*)d_in[1];  // (64,1024)
    const float* w_xh   = (const float*)d_in[2];  // (8192,1024)
    const float* w_hh   = (const float*)d_in[3];  // (1024,1024)
    const float* w_ho   = (const float*)d_in[4];  // (1024,8192)
    const float* b_h    = (const float*)d_in[5];  // (1024,)
    const float* b_o    = (const float*)d_in[6];  // (8192,)

    float* Y       = (float*)d_out;                                   // (256,64,8192)
    float* h_final = (float*)d_out + (long long)SEQ * BATCH * VOCAB;  // (64,1024)

    // Workspace: bf16 transposed weights + ping-pong bf16 hidden state (~18.25 MiB)
    unsigned short* whhT  = (unsigned short*)d_ws;
    unsigned short* whoT  = whhT + (size_t)HIDDEN * HIDDEN;
    unsigned short* hbuf0 = whoT + (size_t)VOCAB * HIDDEN;
    unsigned short* hbuf1 = hbuf0 + (size_t)BATCH * HIDDEN;
    unsigned short* hbuf[2] = { hbuf0, hbuf1 };

    // One-time prep (recomputed every call: deterministic, negligible cost)
    {
        long long n1 = (long long)HIDDEN * HIDDEN;
        long long n2 = (long long)VOCAB * HIDDEN;
        k_transpose_bf16<<<(unsigned)((n1 + 255) / 256), 256, 0, stream>>>(w_hh, whhT, HIDDEN, HIDDEN);
        k_transpose_bf16<<<(unsigned)((n2 + 255) / 256), 256, 0, stream>>>(w_ho, whoT, HIDDEN, VOCAB);
        k_init_h<<<(BATCH * HIDDEN + 255) / 256, 256, 0, stream>>>(status, hbuf0);
    }

    // Fused serial chain: step t computes h_{t+1} (recurrence, blocks 0..3)
    // concurrently with Y[t-1] (projection, blocks 4..35) — both only need h_t.
    for (int t = 0; t < SEQ; ++t) {
        const unsigned short* hin  = hbuf[t & 1];
        unsigned short*       hout = hbuf[(t + 1) & 1];
        float* Yprev = (t == 0) ? nullptr
                                : Y + (long long)(t - 1) * BATCH * VOCAB;
        k_rnn_step<<<36, 256, 0, stream>>>(hin, hout, h_final, whhT, whoT,
                                           X, w_xh, b_h, b_o, Yprev, t);
    }
    // Tail: Y[SEQ-1] from h_SEQ (lives in hbuf[SEQ & 1] == hbuf0)
    k_rnn_project<<<32, 256, 0, stream>>>(hbuf[SEQ & 1], whoT, b_o,
                                          Y + (long long)(SEQ - 1) * BATCH * VOCAB);
}
